// DiffuRNNLayer_32134945308932
// MI455X (gfx1250) — compile-verified, matrix operand order checked by reading, not run on hardware
//
#include <hip/hip_runtime.h>
#include <math.h>

#define B_BATCH 8
#define N_SEQ   2048
#define D_DIM   1024

typedef __attribute__((ext_vector_type(16))) __bf16 v16bf;
typedef __attribute__((ext_vector_type(8)))  __bf16 v8bf;
typedef __attribute__((ext_vector_type(8)))  float  v8f;

// ---------- scalar helpers ----------
__device__ __forceinline__ unsigned short f2bf_bits(float f) {
  unsigned u = __builtin_bit_cast(unsigned, f);
  unsigned r = (u + 0x7FFFu + ((u >> 16) & 1u)) >> 16;  // RNE
  return (unsigned short)r;
}
__device__ __forceinline__ float bf_bits2f(unsigned short s) {
  return __builtin_bit_cast(float, (unsigned)s << 16);
}
__device__ __forceinline__ __bf16 bits2bf(unsigned short s) {
  return __builtin_bit_cast(__bf16, s);
}
__device__ __forceinline__ void split_bf(float a, __bf16& hi, __bf16& lo) {
  unsigned short hs = f2bf_bits(a);
  hi = bits2bf(hs);
  lo = bits2bf(f2bf_bits(a - bf_bits2f(hs)));
}
__device__ __forceinline__ float gelu_f(float x) {
  return 0.5f * x * (1.0f + erff(x * 0.70710678118654752440f));  // exact erf GELU
}

// ---------- one-shot fp32 -> bf16 hi/lo split (same layout) ----------
__global__ void cvt_kernel(const float* __restrict__ in,
                           __bf16* __restrict__ hi, __bf16* __restrict__ lo) {
  long i = (long)blockIdx.x * blockDim.x + threadIdx.x;
  split_bf(in[i], hi[i], lo[i]);
}

// ---------- tiled transpose + split: in [Z,R,C] f32 -> out [Z,C,R] bf16 hi/lo ----------
__global__ void cvtT_kernel(const float* __restrict__ in,
                            __bf16* __restrict__ hi, __bf16* __restrict__ lo,
                            int R, int C) {
  __shared__ float tile[32][33];
  int z = blockIdx.z;
  int r0 = blockIdx.y * 32, c0 = blockIdx.x * 32;
  const float* ib = in + (long)z * R * C;
  #pragma unroll
  for (int t = 0; t < 4; t++) {
    int rr = r0 + threadIdx.y + t * 8;
    tile[threadIdx.y + t * 8][threadIdx.x] = ib[(long)rr * C + c0 + threadIdx.x];
  }
  __syncthreads();
  long ob = (long)z * R * C;
  #pragma unroll
  for (int t = 0; t < 4; t++) {
    int cc = c0 + threadIdx.y + t * 8;               // output row (was column)
    float a = tile[threadIdx.x][threadIdx.y + t * 8];
    long o = ob + (long)cc * R + r0 + threadIdx.x;   // coalesced over r
    split_bf(a, hi[o], lo[o]);
  }
}

// ---------- elementwise depthwise conv stages ----------
// MODE 0: out = x + dt*(conv(x)+cb - x) ; MODE 1: out = gelu(conv+cb) ; MODE 2: out += conv+cb
template<int MODE>
__global__ void dwconv_kernel(const float* __restrict__ in,
                              const float* __restrict__ w,
                              const float* __restrict__ cb,
                              const float* __restrict__ dtp,
                              float* __restrict__ out) {
  long idx = (long)blockIdx.x * blockDim.x + threadIdx.x;
  int  d  = (int)(idx % D_DIM);
  long bn = idx / D_DIM;
  int  n  = (int)(bn % N_SEQ);
  long b  = bn / N_SEQ;
  const float* row = in + b * (long)N_SEQ * D_DIM + d;
  float xm1 = (n > 0)         ? row[(long)(n - 1) * D_DIM] : 0.0f;
  float x0  =                   row[(long)n       * D_DIM];
  float xp1 = (n < N_SEQ - 1) ? row[(long)(n + 1) * D_DIM] : 0.0f;
  float y = w[d * 3 + 0] * xm1 + w[d * 3 + 1] * x0 + w[d * 3 + 2] * xp1 + cb[d];
  if constexpr (MODE == 0)      out[idx] = x0 + dtp[0] * (y - x0);
  else if constexpr (MODE == 1) out[idx] = gelu_f(y);
  else                          out[idx] += y;
}

// ---------- LayerNorm: one 256-thread block per row of D=1024 ----------
__global__ void ln_kernel(const float* __restrict__ in,
                          const float* __restrict__ g,
                          const float* __restrict__ be,
                          float* __restrict__ out) {
  long row = blockIdx.x;
  const float* p = in + row * (long)D_DIM;
  float s = 0.f, s2 = 0.f;
  for (int i = threadIdx.x; i < D_DIM; i += 256) { float v = p[i]; s += v; s2 += v * v; }
  __shared__ float shs[8], shs2[8];
  #pragma unroll
  for (int o = 16; o > 0; o >>= 1) { s += __shfl_down(s, o, 32); s2 += __shfl_down(s2, o, 32); }
  int wid = threadIdx.x >> 5;
  if ((threadIdx.x & 31) == 0) { shs[wid] = s; shs2[wid] = s2; }
  __syncthreads();
  if (threadIdx.x == 0) {
    float a = 0.f, b2 = 0.f;
    #pragma unroll
    for (int i = 0; i < 8; i++) { a += shs[i]; b2 += shs2[i]; }
    float mean = a * (1.0f / D_DIM);
    shs[0] = mean;
    shs2[0] = b2 * (1.0f / D_DIM) - mean * mean;
  }
  __syncthreads();
  float mean = shs[0];
  float inv = rsqrtf(shs2[0] + 1e-5f);
  for (int i = threadIdx.x; i < D_DIM; i += 256)
    out[row * (long)D_DIM + i] = (p[i] - mean) * inv * g[i] + be[i];
}

// ---------- K_sum[b,d] = sum_n Kp[b,n,d] ----------
__global__ void ksum_kernel(const float* __restrict__ kp, float* __restrict__ ksum) {
  int d = blockIdx.x * blockDim.x + threadIdx.x;
  int b = blockIdx.y;
  const float* p = kp + (long)b * N_SEQ * D_DIM + d;
  float s = 0.f;
  for (int n = 0; n < N_SEQ; n++) s += p[(long)n * D_DIM];
  ksum[b * D_DIM + d] = s;
}

// ---------- norm[b,n] = dot(Qp[b,n,:], Ksum[b,:]) ; one wave per row ----------
__global__ void norm_kernel(const float* __restrict__ qp,
                            const float* __restrict__ ksum,
                            float* __restrict__ nrm) {
  long row = (long)blockIdx.x * 8 + (threadIdx.x >> 5);
  int lane = threadIdx.x & 31;
  long b = row / N_SEQ;
  const float* q  = qp + row * (long)D_DIM;
  const float* ks = ksum + b * D_DIM;
  float s = 0.f;
  for (int i = lane; i < D_DIM; i += 32) s += q[i] * ks[i];
  #pragma unroll
  for (int o = 16; o > 0; o >>= 1) s += __shfl_down(s, o, 32);
  if (lane == 0) nrm[row] = s;
}

// ---------- bf16x3 WMMA GEMM on pre-split bf16 planes ----------
// All operands contiguous in K: A[M,K] row-major, B[N,K] ("weight layout").
// Fragment: 16 halves/lane; elems 0-7 <- k = kb+h*8.., elems 8-15 <- k = kb+16+h*8..
__device__ __forceinline__ v16bf load_frag(const __bf16* __restrict__ base, int ld,
                                           int rr, int kBase, int h) {
  const __bf16* p = base + (long)rr * ld + kBase + h * 8;
  union { v16bf v; v8bf q[2]; } u;
  u.q[0] = *(const v8bf*)(p);        // 16B global_load_b128
  u.q[1] = *(const v8bf*)(p + 16);   // 16B global_load_b128
  return u.v;
}

#define WMMA_BF16(d, a, b) \
  d = __builtin_amdgcn_wmma_f32_16x16x32_bf16(false, (a), false, (b), (short)0, (d), false, false)

// ACT: 0 none, 1 exact GELU, 2 elu(x)+1.
// CMB: 0 store, 1 C += r, 2 C += r/(extra[m]+1e-6), 3 C = r + extra[m*ldc+n].
template<int ACT, int CMB>
__global__ __launch_bounds__(256, 1)
void gemm_wmma_kernel(const __bf16* __restrict__ Ah, const __bf16* __restrict__ Al,
                      long sA, int lda,
                      const __bf16* __restrict__ Bh, const __bf16* __restrict__ Bl,
                      long sB, int ldb,
                      const float* __restrict__ bias,
                      const float* __restrict__ extra, long sE,
                      float* __restrict__ C, long sC, int ldc,
                      int K) {
  const __bf16* Ahb = Ah + (long)blockIdx.z * sA;
  const __bf16* Alb = Al + (long)blockIdx.z * sA;
  const __bf16* Bhb = Bh + (long)blockIdx.z * sB;
  const __bf16* Blb = Bl + (long)blockIdx.z * sB;
  float*        Cb  = C  + (long)blockIdx.z * sC;
  const float*  Eb  = extra ? (extra + (long)blockIdx.z * sE) : nullptr;

  int wid  = threadIdx.x >> 5;
  int lane = threadIdx.x & 31;
  int r = lane & 15, h = lane >> 4;
  int mBase = blockIdx.x * 128 + (wid & 3) * 32;   // wave: 32 rows
  int nBase = blockIdx.y * 128 + (wid >> 2) * 64;  // wave: 64 cols

  const v8f zero = {0.f, 0.f, 0.f, 0.f, 0.f, 0.f, 0.f, 0.f};
  v8f acc[2][4];
  #pragma unroll
  for (int i = 0; i < 2; i++)
    #pragma unroll
    for (int j = 0; j < 4; j++) acc[i][j] = zero;

  for (int kb = 0; kb < K; kb += 32) {
    if (kb + 32 < K) {  // global_prefetch_b8 next K tile
      __builtin_prefetch(&Ahb[(long)(mBase + r) * lda + kb + 32], 0, 1);
      __builtin_prefetch(&Bhb[(long)(nBase + r) * ldb + kb + 32], 0, 1);
    }
    v16bf a0h = load_frag(Ahb, lda, mBase + r,      kb, h);
    v16bf a0l = load_frag(Alb, lda, mBase + r,      kb, h);
    v16bf a1h = load_frag(Ahb, lda, mBase + 16 + r, kb, h);
    v16bf a1l = load_frag(Alb, lda, mBase + 16 + r, kb, h);
    #pragma unroll
    for (int j = 0; j < 4; j++) {
      v16bf bh = load_frag(Bhb, ldb, nBase + 16 * j + r, kb, h);
      v16bf bl = load_frag(Blb, ldb, nBase + 16 * j + r, kb, h);
      // bf16x3: hi*hi + hi*lo + lo*hi (lo*lo below f32 rounding)
      WMMA_BF16(acc[0][j], a0h, bh);
      WMMA_BF16(acc[0][j], a0h, bl);
      WMMA_BF16(acc[0][j], a0l, bh);
      WMMA_BF16(acc[1][j], a1h, bh);
      WMMA_BF16(acc[1][j], a1h, bl);
      WMMA_BF16(acc[1][j], a1l, bh);
    }
  }

  // epilogue: C layout -> m = v + 8*h (+16*i), n = r (+16*j)
  #pragma unroll
  for (int i = 0; i < 2; i++) {
    #pragma unroll
    for (int j = 0; j < 4; j++) {
      #pragma unroll
      for (int v = 0; v < 8; v++) {
        int m = mBase + i * 16 + v + 8 * h;
        int n = nBase + j * 16 + r;
        float val = acc[i][j][v];
        if (bias) val += bias[n];
        if constexpr (ACT == 1) val = gelu_f(val);
        else if constexpr (ACT == 2) val = (val > 0.f) ? (val + 1.0f) : expf(val);
        long off = (long)m * ldc + n;
        if constexpr (CMB == 0)      Cb[off] = val;
        else if constexpr (CMB == 1) Cb[off] += val;
        else if constexpr (CMB == 2) Cb[off] += val / (Eb[m] + 1e-6f);
        else                         Cb[off] = val + Eb[off];
      }
    }
  }
}

// ---------- launch ----------
extern "C" void kernel_launch(void* const* d_in, const int* in_sizes, int n_in,
                              void* d_out, int out_size, void* d_ws, size_t ws_size,
                              hipStream_t stream) {
  (void)in_sizes; (void)n_in; (void)out_size; (void)ws_size;
  const float* x      = (const float*)d_in[0];
  const float* dt     = (const float*)d_in[1];
  const float* diff_w = (const float*)d_in[2];
  const float* diff_b = (const float*)d_in[3];
  const float* lu_w1  = (const float*)d_in[4];
  const float* lu_b1  = (const float*)d_in[5];
  const float* lu_w2  = (const float*)d_in[6];
  const float* lu_b2  = (const float*)d_in[7];
  const float* tm_g   = (const float*)d_in[8];
  const float* tm_be  = (const float*)d_in[9];
  const float* tm_w1  = (const float*)d_in[10];
  const float* tm_cb1 = (const float*)d_in[11];
  const float* tm_w2  = (const float*)d_in[12];
  const float* tm_cb2 = (const float*)d_in[13];
  const float* wq     = (const float*)d_in[14];
  const float* bq     = (const float*)d_in[15];
  const float* wk     = (const float*)d_in[16];
  const float* bk     = (const float*)d_in[17];
  const float* wv     = (const float*)d_in[18];
  const float* bv     = (const float*)d_in[19];
  const float* ff_w1  = (const float*)d_in[20];
  const float* ff_b1  = (const float*)d_in[21];
  const float* ff_w2  = (const float*)d_in[22];
  const float* ff_b2  = (const float*)d_in[23];
  const float* n1_g   = (const float*)d_in[24];
  const float* n1_b   = (const float*)d_in[25];
  const float* n2_g   = (const float*)d_in[26];
  const float* n2_b   = (const float*)d_in[27];
  float* out = (float*)d_out;

  const long BND = (long)B_BATCH * N_SEQ * D_DIM;   // 16,777,216
  const long DD  = (long)D_DIM * D_DIM;             // 1,048,576
  const long BDD = (long)B_BATCH * DD;              // 8,388,608

  float* ws   = (float*)d_ws;
  float* acc  = ws + 0 * BND;
  float* lnb  = ws + 1 * BND;
  float* t1v  = ws + 2 * BND;   // token-mixer mid; later V; later pre-LN2
  float* hbuf = ws + 3 * BND;
  float* qp   = ws + 4 * BND;
  float* kp   = ws + 5 * BND;
  float* kv   = ws + 6 * BND;                    // [B,D,D]
  float* ksum = kv + BDD;                        // [B,D]
  float* nrm  = ksum + B_BATCH * D_DIM;          // [B,N]

  __bf16* bf   = (__bf16*)(nrm + B_BATCH * N_SEQ);  // 16B-aligned (offsets all mult. of 4 floats)
  __bf16* xh   = bf;             __bf16* xl   = xh   + BND;   // x planes (reused by 4 GEMMs)
  __bf16* gh   = xl   + BND;     __bf16* gl   = gh   + BND;   // generic activation planes
  __bf16* kpTh = gl   + BND;     __bf16* kpTl = kpTh + BND;   // Kp^T [B,D,N]
  __bf16* vTh  = kpTl + BND;     __bf16* vTl  = vTh  + BND;   // V^T  [B,D,N]
  __bf16* kvTh = vTl  + BND;     __bf16* kvTl = kvTh + BDD;   // KV^T [B,D,D]
  __bf16* wh   = kvTl + BDD;     __bf16* wl   = wh   + 7 * DD; // 7 weight planes

  const dim3 blk(256);
  const int  EW  = (int)(BND / 256);
  const int  WW  = (int)(DD / 256);
  const dim3 gFull(128, 8, 1);
  const dim3 tblk(32, 8, 1);
  const long sBN = (long)N_SEQ * D_DIM;

  // weight splits (once; constant per call)
  const float* wsrc[7] = {lu_w1, lu_w2, wq, wk, wv, ff_w1, ff_w2};
  for (int i = 0; i < 7; i++)
    cvt_kernel<<<WW, blk, 0, stream>>>(wsrc[i], wh + i * DD, wl + i * DD);

  // 1) diffusion + 3) token mixer (bandwidth stages)
  dwconv_kernel<0><<<EW, blk, 0, stream>>>(x, diff_w, diff_b, dt, acc);
  ln_kernel<<<(int)(BND / D_DIM), blk, 0, stream>>>(x, tm_g, tm_be, lnb);
  dwconv_kernel<1><<<EW, blk, 0, stream>>>(lnb, tm_w1, tm_cb1, nullptr, t1v);
  dwconv_kernel<2><<<EW, blk, 0, stream>>>(t1v, tm_w2, tm_cb2, nullptr, acc);

  // split x once; feeds 4 GEMMs
  cvt_kernel<<<EW, blk, 0, stream>>>(x, xh, xl);

  // 2) local MLP
  gemm_wmma_kernel<1, 0><<<gFull, blk, 0, stream>>>(
      xh, xl, 0, D_DIM, wh + 0 * DD, wl + 0 * DD, 0, D_DIM,
      lu_b1, nullptr, 0, hbuf, 0, D_DIM, D_DIM);
  cvt_kernel<<<EW, blk, 0, stream>>>(hbuf, gh, gl);
  gemm_wmma_kernel<0, 1><<<gFull, blk, 0, stream>>>(
      gh, gl, 0, D_DIM, wh + 1 * DD, wl + 1 * DD, 0, D_DIM,
      lu_b2, nullptr, 0, acc, 0, D_DIM, D_DIM);

  // 4) linear attention projections
  gemm_wmma_kernel<2, 0><<<gFull, blk, 0, stream>>>(
      xh, xl, 0, D_DIM, wh + 2 * DD, wl + 2 * DD, 0, D_DIM,
      bq, nullptr, 0, qp, 0, D_DIM, D_DIM);
  gemm_wmma_kernel<2, 0><<<gFull, blk, 0, stream>>>(
      xh, xl, 0, D_DIM, wh + 3 * DD, wl + 3 * DD, 0, D_DIM,
      bk, nullptr, 0, kp, 0, D_DIM, D_DIM);
  gemm_wmma_kernel<0, 0><<<gFull, blk, 0, stream>>>(
      xh, xl, 0, D_DIM, wh + 4 * DD, wl + 4 * DD, 0, D_DIM,
      bv, nullptr, 0, t1v, 0, D_DIM, D_DIM);

  ksum_kernel<<<dim3(D_DIM / 256, B_BATCH), blk, 0, stream>>>(kp, ksum);
  norm_kernel<<<(int)(BND / D_DIM / 8), blk, 0, stream>>>(qp, ksum, nrm);

  // transpose-split Kp, V -> [B,D,N] planes
  cvtT_kernel<<<dim3(D_DIM / 32, N_SEQ / 32, B_BATCH), tblk, 0, stream>>>(kp, kpTh, kpTl, N_SEQ, D_DIM);
  cvtT_kernel<<<dim3(D_DIM / 32, N_SEQ / 32, B_BATCH), tblk, 0, stream>>>(t1v, vTh, vTl, N_SEQ, D_DIM);

  // KV[b] = Kp[b]^T @ V[b]  (M=D, N=D, K=N_SEQ; B operand = V^T in weight layout)
  gemm_wmma_kernel<0, 0><<<dim3(8, 8, B_BATCH), blk, 0, stream>>>(
      kpTh, kpTl, (long)D_DIM * N_SEQ, N_SEQ,
      vTh, vTl, (long)D_DIM * N_SEQ, N_SEQ,
      nullptr, nullptr, 0, kv, DD, D_DIM, N_SEQ);

  // KV^T planes; then acc += (Qp @ KV) / (norm + 1e-6)
  cvtT_kernel<<<dim3(D_DIM / 32, D_DIM / 32, B_BATCH), tblk, 0, stream>>>(kv, kvTh, kvTl, D_DIM, D_DIM);
  cvt_kernel<<<EW, blk, 0, stream>>>(qp, gh, gl);
  gemm_wmma_kernel<0, 2><<<dim3(16, 8, B_BATCH), blk, 0, stream>>>(
      gh, gl, sBN, D_DIM, kvTh, kvTl, DD, D_DIM,
      nullptr, nrm, N_SEQ, acc, sBN, D_DIM, D_DIM);

  // 5) LN1 -> FFN (residual) -> LN2
  ln_kernel<<<(int)(BND / D_DIM), blk, 0, stream>>>(acc, n1_g, n1_b, lnb);
  cvt_kernel<<<EW, blk, 0, stream>>>(lnb, gh, gl);
  gemm_wmma_kernel<1, 0><<<gFull, blk, 0, stream>>>(
      gh, gl, 0, D_DIM, wh + 5 * DD, wl + 5 * DD, 0, D_DIM,
      ff_b1, nullptr, 0, hbuf, 0, D_DIM, D_DIM);
  cvt_kernel<<<EW, blk, 0, stream>>>(hbuf, gh, gl);
  gemm_wmma_kernel<0, 3><<<gFull, blk, 0, stream>>>(
      gh, gl, 0, D_DIM, wh + 6 * DD, wl + 6 * DD, 0, D_DIM,
      ff_b2, lnb, 0, t1v, 0, D_DIM, D_DIM);
  ln_kernel<<<(int)(BND / D_DIM), blk, 0, stream>>>(t1v, n2_g, n2_b, out);
}
